// RNN_37692632990219
// MI455X (gfx1250) — matrix operand order repaired and checked by reading, not compiled, not run on hardware
//
#include <hip/hip_runtime.h>
#include <cstdint>
#include <cstddef>

// ============================================================================
// 2-layer GRU (B=64, T=1024, D=256, H=512) + dense head, for MI455X (gfx1250).
//
//   K1  gemm_xg:  xg = A(bf16) @ W_i^T(bf16) + b_i   -> f32 [65536][1536]
//       WMMA bf16 16x16x32, 128x128 block tiles. LDS tiles are filled by the
//       Tensor Data Mover (tensor_load_to_lds, TENSORcnt), double buffered;
//       TDM pad fields reproduce the stride-40 (bf16) anti-bank-conflict pad.
//   K2  gru_scan: persistent 32-WG kernel, 1024 sequential steps.
//       4 batch groups x 8 hidden slices. W_h slice (192x512 bf16, ~200KB
//       padded) resident in the 320KB WGP LDS, loaded by one 3-D TDM op.
//       f32 hidden state lives in registers (each lane owns its columns);
//       only the bf16 copy is exchanged via global + per-step 16x512 TDM.
//       Per-step sync = per-group atomic arrival counter + agent fences.
//   K3  dense_out: out = tanh(tanh(c1) @ W_out + b_out), 128 WMMA tiles.
// ============================================================================

typedef __bf16 v16bf __attribute__((ext_vector_type(16)));
typedef __bf16 v8bf  __attribute__((ext_vector_type(8)));
typedef float  v8f   __attribute__((ext_vector_type(8)));
typedef unsigned int v4u __attribute__((ext_vector_type(4)));
typedef unsigned int v8u __attribute__((ext_vector_type(8)));

#define DEV __device__ __forceinline__

DEV unsigned short f2bf(float f) {  // round-to-nearest-even f32 -> bf16 bits
  union { float f; unsigned u; } v; v.f = f;
  unsigned r = v.u + 0x7FFFu + ((v.u >> 16) & 1u);
  return (unsigned short)(r >> 16);
}

DEV v16bf cat16(v8bf lo, v8bf hi) {
  return __builtin_shufflevector(lo, hi, 0,1,2,3,4,5,6,7,8,9,10,11,12,13,14,15);
}

// A fragment (16x32, bf16): lane holds row M = lane&15.
// lanes 0-15 hold K {k0..k0+7, k0+16..k0+23}; lanes 16-31 hold {+8, +24}.
DEV v16bf ldA(const unsigned short* row_k0, int hi) {
  v8bf lo  = *(const v8bf*)(row_k0 + 8 * hi);
  v8bf hi8 = *(const v8bf*)(row_k0 + 16 + 8 * hi);
  return cat16(lo, hi8);
}
// B fragment (32x16, bf16, stored [N][K]): lane holds col N = lane&15,
// K {k0..k0+7, k0+16..k0+23} for lanes 0-15; {+8, +24} for lanes 16-31.
DEV v16bf ldB(const unsigned short* col_k0, int hi) {
  v8bf lo  = *(const v8bf*)(col_k0 + 8 * hi);
  v8bf hi8 = *(const v8bf*)(col_k0 + 16 + 8 * hi);
  return cat16(lo, hi8);
}
DEV v8f wmma_bf16(v16bf a, v16bf b, v8f c) {
  return __builtin_amdgcn_wmma_f32_16x16x32_bf16(false, a, false, b,
                                                 (short)0, c, false, false);
}
DEV float sigmoidf_(float x) { return 1.f / (1.f + __expf(-x)); }

// ---------------------------------------------------------------------------
// Tensor Data Mover: DMA a [dim2 x rows x row_elems] bf16 tile into LDS with
// per-row padding (pad_amount = 4 DWORDs = 8 bf16).  D# per CDNA5 ISA.
//   pad_interval_code: 3 -> pad every 16 DWORDs (row of 32 bf16, stride 40)
//                      7 -> pad every 256 DWORDs (row of 512 bf16, stride 520)
// Issue from ONE wave only; completion via s_wait_tensorcnt + barrier.
// ---------------------------------------------------------------------------
DEV void tdm_load(void* lds_dst, const void* gsrc, unsigned rows,
                  unsigned row_elems, unsigned long long stride0_elems,
                  unsigned pad_interval_code, unsigned dim2,
                  unsigned long long stride1_elems) {
  const unsigned long long ga = (unsigned long long)(size_t)gsrc;
  v4u g0;
  g0.x = 1u;                                   // count=1 (valid user D#)
  g0.y = (unsigned)(size_t)lds_dst;            // lds_addr (bytes)
  g0.z = (unsigned)ga;                         // global_addr[31:0]
  g0.w = (unsigned)((ga >> 32) & 0x01FFFFFFu)  // global_addr[56:32]
       | (2u << 30);                           // type = 2 ("image")
  v8u g1;
  g1.s0 = (1u << 16)                           // data_size = 1 (2 bytes)
        | (1u << 20)                           // pad_enable
        | (pad_interval_code << 22)            // pad_interval
        | (3u << 25);                          // pad_amount = 4 DWORDs
  g1.s1 = (row_elems & 0xFFFFu) << 16;         // tensor_dim0 [15:0] @ bit48
  g1.s2 = (row_elems >> 16) | ((rows & 0xFFFFu) << 16);   // dim0 hi | dim1 lo
  g1.s3 = (rows >> 16) | ((row_elems & 0xFFFFu) << 16);   // dim1 hi | tile_dim0
  g1.s4 = (rows & 0xFFFFu) | (dim2 << 16);     // tile_dim1 | tile_dim2
  g1.s5 = (unsigned)stride0_elems;             // tensor_dim0_stride lo32
  g1.s6 = (unsigned)(stride0_elems >> 32)
        | ((unsigned)(stride1_elems & 0xFFFFu) << 16);    // dim1_stride lo16
  g1.s7 = (unsigned)(stride1_elems >> 16);     // dim1_stride [47:16]
  v4u g2; g2.x = dim2; g2.y = 0u; g2.z = 0u; g2.w = 0u;   // tensor_dim2
  v4u g3; g3.x = 0u;  g3.y = 0u; g3.z = 0u; g3.w = 0u;
  asm volatile("tensor_load_to_lds %0, %1, %2, %3"
               :: "s"(g0), "s"(g1), "s"(g2), "s"(g3)
               : "memory");
}
DEV void wait_tensor0() { __builtin_amdgcn_s_wait_tensorcnt(0); }

// ---------------------------------------------------------------------------
// Small prep kernels
// ---------------------------------------------------------------------------
__global__ void f32_to_bf16_kernel(const float* __restrict__ src,
                                   unsigned short* __restrict__ dst, int n) {
  int i = blockIdx.x * blockDim.x + threadIdx.x;
  if (i < n) dst[i] = f2bf(src[i]);
}

// W [K][N] f32  ->  WT [N][K] bf16  (write-contiguous)
__global__ void wtrans_kernel(const float* __restrict__ W,
                              unsigned short* __restrict__ WT, int K, int N) {
  int i = blockIdx.x * blockDim.x + threadIdx.x;
  if (i >= K * N) return;
  int n = i / K, k = i - n * K;
  WT[i] = f2bf(W[(size_t)k * N + n]);
}

__global__ void zero_kernel(unsigned int* __restrict__ p, int n) {
  int i = blockIdx.x * blockDim.x + threadIdx.x;
  if (i < n) p[i] = 0u;
}

// ---------------------------------------------------------------------------
// K1: xg = A @ Bt^T + bias.  A: [65536][K] bf16, Bt: [1536][K] bf16,
// C: [65536][1536] f32.  Block tile 128x128, 8 waves of 64x32.
// LDS tiles (stride 40 bf16) are DMA'd by wave 0 via TDM, double buffered.
// ---------------------------------------------------------------------------
__global__ __launch_bounds__(256) void gemm_xg_kernel(
    const unsigned short* __restrict__ A,
    const unsigned short* __restrict__ Bt,
    const float* __restrict__ bias,
    float* __restrict__ C,
    int K) {
  constexpr int N = 1536;
  __shared__ unsigned short As[2][128 * 40];
  __shared__ unsigned short Bs[2][128 * 40];

  const int tid = threadIdx.x;
  const int lane = tid & 31, wv = tid >> 5;
  const int wm = wv >> 2, wn = wv & 3;       // 2x4 wave grid
  const int lr = lane & 15, hi = lane >> 4;
  const int m0 = blockIdx.x * 128;
  const int n0 = blockIdx.y * 128;

  v8f acc[4][2] = {};
  const int nk = K >> 5;

  if (wv == 0) {
    tdm_load(&As[0][0], A + (size_t)m0 * K, 128, 32, (unsigned)K, 3, 0, 0);
    tdm_load(&Bs[0][0], Bt + (size_t)n0 * K, 128, 32, (unsigned)K, 3, 0, 0);
  }
  for (int kb = 0; kb < nk; ++kb) {
    const int buf = kb & 1;
    if (wv == 0) wait_tensor0();
    __syncthreads();
    if (wv == 0 && kb + 1 < nk) {   // DMA next tile while 8 waves compute
      tdm_load(&As[buf ^ 1][0], A + (size_t)m0 * K + (kb + 1) * 32,
               128, 32, (unsigned)K, 3, 0, 0);
      tdm_load(&Bs[buf ^ 1][0], Bt + (size_t)n0 * K + (kb + 1) * 32,
               128, 32, (unsigned)K, 3, 0, 0);
    }
    v16bf af[4], bfr[2];
#pragma unroll
    for (int mt = 0; mt < 4; ++mt)
      af[mt] = ldA(&As[buf][(wm * 64 + mt * 16 + lr) * 40], hi);
#pragma unroll
    for (int nt = 0; nt < 2; ++nt)
      bfr[nt] = ldB(&Bs[buf][(wn * 32 + nt * 16 + lr) * 40], hi);
#pragma unroll
    for (int mt = 0; mt < 4; ++mt)
#pragma unroll
      for (int nt = 0; nt < 2; ++nt)
        acc[mt][nt] = wmma_bf16(af[mt], bfr[nt], acc[mt][nt]);
  }

#pragma unroll
  for (int mt = 0; mt < 4; ++mt) {
    const int rowb = m0 + wm * 64 + mt * 16 + 8 * hi;
#pragma unroll
    for (int nt = 0; nt < 2; ++nt) {
      const int col = n0 + wn * 32 + nt * 16 + lr;
      const float bia = bias[col];
#pragma unroll
      for (int r = 0; r < 8; ++r)
        C[(size_t)(rowb + r) * N + col] = acc[mt][nt][r] + bia;
    }
  }
}

// ---------------------------------------------------------------------------
// K2: persistent GRU scan for one layer.
// grid = 32 WGs x 128 threads.  blockIdx: g = >>3 (batch group of 16 rows),
// s = &7 (hidden slice of 64 cols).  Wave wv owns cols [s*64+wv*16, +16) of
// ALL THREE gates -> gate math is lane-local; f32 h state stays in registers.
// ---------------------------------------------------------------------------
__global__ __launch_bounds__(128) void gru_scan_kernel(
    const float* __restrict__ xg,            // [64*T][1536], row = b*T + t
    const unsigned short* __restrict__ WhT,  // [1536][512] bf16
    const float* __restrict__ b_hn,          // [512]
    unsigned short* __restrict__ hbf,        // [2][64][512] bf16 (pre-zeroed)
    unsigned short* __restrict__ ybf,        // [64*T][512] bf16 tanh(h) | null
    float* __restrict__ c_out,               // [64][512] final hidden
    unsigned short* __restrict__ ct_out,     // [64][512] bf16 tanh(final)|null
    unsigned int* __restrict__ ctr,          // [4] arrival counters (zeroed)
    int T) {
  extern __shared__ unsigned short sm[];
  unsigned short* Ws = sm;               // [192][520] resident W_h slice
  unsigned short* hA = sm + 192 * 520;   // [16][520]  h_prev bf16 A-tile

  const int tid = threadIdx.x;
  const int lane = tid & 31, wv = tid >> 5;
  const int lr = lane & 15, hi = lane >> 4;
  const int g = blockIdx.x >> 3;
  const int s = blockIdx.x & 7;

  if (wv == 0) {
    // one 3-D TDM op: 3 gate blocks x 64 rows x 512 elems; LDS stride 520
    tdm_load(Ws, WhT + (size_t)(s * 64) * 512, 64, 512, 512, 7,
             /*dim2=*/3, /*stride1=*/512ull * 512ull);
    wait_tensor0();
  }
  const int j = s * 64 + wv * 16 + lr;  // this lane's hidden column
  const float bhn = b_hn[j];
  float hreg[8];                         // f32 state for rows m = 8*hi + r
#pragma unroll
  for (int r = 0; r < 8; ++r) hreg[r] = 0.f;
  __syncthreads();

  for (int t = 0; t < T; ++t) {
    const int cur = t & 1, nxt = cur ^ 1;

    // DMA h_prev (16 rows x 512 bf16) for this batch group into LDS
    if (wv == 0) {
      tdm_load(hA, hbf + cur * (64 * 512) + g * (16 * 512), 16, 512, 512, 7,
               0, 0);
      wait_tensor0();
    }
    __syncthreads();

    // hg tile: [16 x 48] per wave = 3 gate tiles, K = 512
    v8f acc[3] = {};
#pragma unroll 4
    for (int kk = 0; kk < 16; ++kk) {
      const int k0 = kk * 32;
      const v16bf af = ldA(hA + lr * 520 + k0, hi);
#pragma unroll
      for (int gt = 0; gt < 3; ++gt) {
        const v16bf bfr = ldB(Ws + (gt * 64 + wv * 16 + lr) * 520 + k0, hi);
        acc[gt] = wmma_bf16(af, bfr, acc[gt]);
      }
    }

    // gates: lane owns column j for 8 batch rows (m = 8*hi + r)
#pragma unroll
    for (int r = 0; r < 8; ++r) {
      const int b = g * 16 + 8 * hi + r;
      const size_t xrow = ((size_t)b * T + t) * 1536;
      const float xr = xg[xrow + j];
      const float xz = xg[xrow + 512 + j];
      const float xn = xg[xrow + 1024 + j];
      if (t + 1 < T) __builtin_prefetch(xg + xrow + 1536 + j, 0, 1);
      const float rg = sigmoidf_(xr + acc[0][r]);
      const float zg = sigmoidf_(xz + acc[1][r]);
      const float ng = tanhf(xn + rg * (acc[2][r] + bhn));
      const float hnew = (1.f - zg) * ng + zg * hreg[r];
      hreg[r] = hnew;
      hbf[nxt * (64 * 512) + b * 512 + j] = f2bf(hnew);
      if (ybf) ybf[((size_t)b * T + t) * 512 + j] = f2bf(tanhf(hnew));
      if (t == T - 1) {
        c_out[b * 512 + j] = hnew;
        if (ct_out) ct_out[b * 512 + j] = f2bf(tanhf(hnew));
      }
    }

    // 8-WG barrier within batch group g (monotonic counter, agent scope)
    __builtin_amdgcn_fence(__ATOMIC_RELEASE, "agent");
    __syncthreads();
    if (tid == 0) {
      __hip_atomic_fetch_add(&ctr[g], 1u, __ATOMIC_RELAXED,
                             __HIP_MEMORY_SCOPE_AGENT);
      const unsigned int tgt = 8u * (unsigned)(t + 1);
      while (__hip_atomic_load(&ctr[g], __ATOMIC_RELAXED,
                               __HIP_MEMORY_SCOPE_AGENT) < tgt)
        __builtin_amdgcn_s_sleep(1);
    }
    __syncthreads();
    __builtin_amdgcn_fence(__ATOMIC_ACQUIRE, "agent");
  }
}

// ---------------------------------------------------------------------------
// K3: out = tanh( tanh(c1) @ W_out + b_out ).  128 tiles, 1 wave each.
// ---------------------------------------------------------------------------
__global__ __launch_bounds__(128) void dense_out_kernel(
    const unsigned short* __restrict__ c1t,  // [64][512] bf16 tanh(c1)
    const unsigned short* __restrict__ WoT,  // [512][512] bf16 [N][K]
    const float* __restrict__ b_out,         // [512]
    float* __restrict__ out) {               // [64][512]
  const int tid = threadIdx.x, lane = tid & 31, wv = tid >> 5;
  const int lr = lane & 15, hi = lane >> 4;
  const int id = blockIdx.x * 4 + wv;  // 0..127
  const int mt = id >> 5, nt = id & 31;
  v8f acc = {};
  for (int kk = 0; kk < 16; ++kk) {
    const int k0 = kk * 32;
    const v16bf af = ldA(c1t + (size_t)(mt * 16 + lr) * 512 + k0, hi);
    const v16bf bfr = ldB(WoT + (size_t)(nt * 16 + lr) * 512 + k0, hi);
    acc = wmma_bf16(af, bfr, acc);
  }
  const int n = nt * 16 + lr;
  const float bia = b_out[n];
#pragma unroll
  for (int r = 0; r < 8; ++r)
    out[(mt * 16 + 8 * hi + r) * 512 + n] = tanhf(acc[r] + bia);
}

// ---------------------------------------------------------------------------
// Host side
// ---------------------------------------------------------------------------
extern "C" void kernel_launch(void* const* d_in, const int* in_sizes, int n_in,
                              void* d_out, int out_size, void* d_ws,
                              size_t ws_size, hipStream_t stream) {
  (void)in_sizes; (void)n_in; (void)out_size; (void)ws_size;
  const float* x     = (const float*)d_in[0];
  const float* W_i0  = (const float*)d_in[1];
  const float* b_i0  = (const float*)d_in[2];
  const float* W_h0  = (const float*)d_in[3];
  const float* b_hn0 = (const float*)d_in[4];
  const float* W_i1  = (const float*)d_in[5];
  const float* b_i1  = (const float*)d_in[6];
  const float* W_h1  = (const float*)d_in[7];
  const float* b_hn1 = (const float*)d_in[8];
  const float* W_out = (const float*)d_in[9];
  const float* b_out = (const float*)d_in[10];
  float* out = (float*)d_out;  // [out | c0 | c1], 3 x 64*512 f32

  char* ws = (char*)d_ws;
  size_t off = 0;
  auto carve = [&](size_t bytes) {
    size_t o = off; off += (bytes + 255) & ~(size_t)255; return o;
  };
  const size_t oXG   = carve((size_t)65536 * 1536 * 4);  // xg f32 (both layers)
  const size_t oXB   = carve((size_t)65536 * 256 * 2);   // x bf16
  const size_t oYB   = carve((size_t)65536 * 512 * 2);   // y0 bf16
  const size_t oWIT0 = carve((size_t)1536 * 256 * 2);
  const size_t oWHT0 = carve((size_t)1536 * 512 * 2);
  const size_t oWIT1 = carve((size_t)1536 * 512 * 2);
  const size_t oWHT1 = carve((size_t)1536 * 512 * 2);
  const size_t oWOT  = carve((size_t)512 * 512 * 2);
  const size_t oC1T  = carve((size_t)64 * 512 * 2);
  const size_t oHBF  = carve((size_t)2 * 2 * 64 * 512 * 2);  // bf16 h x2 layers
  const size_t oCTR  = carve(256);                           // ctr0|ctr1

  unsigned short* xb   = (unsigned short*)(ws + oXB);
  unsigned short* yb   = (unsigned short*)(ws + oYB);
  unsigned short* WiT0 = (unsigned short*)(ws + oWIT0);
  unsigned short* WhT0 = (unsigned short*)(ws + oWHT0);
  unsigned short* WiT1 = (unsigned short*)(ws + oWIT1);
  unsigned short* WhT1 = (unsigned short*)(ws + oWHT1);
  unsigned short* WoT  = (unsigned short*)(ws + oWOT);
  unsigned short* c1t  = (unsigned short*)(ws + oC1T);
  float* xg = (float*)(ws + oXG);
  unsigned short* hbf0 = (unsigned short*)(ws + oHBF);
  unsigned short* hbf1 = hbf0 + 2 * 64 * 512;
  unsigned int* ctr0 = (unsigned int*)(ws + oCTR);
  unsigned int* ctr1 = ctr0 + 32;

  constexpr int SCAN_LDS = (192 * 520 + 16 * 520) * 2;  // 216,320 B < 320 KB
  (void)hipFuncSetAttribute((const void*)gru_scan_kernel,
                            hipFuncAttributeMaxDynamicSharedMemorySize,
                            SCAN_LDS);

  // 0) zero bf16 h state + barrier counters (contiguous region)
  {
    const int nw = (2 * 2 * 64 * 512 * 2) / 4 + 64;  // words
    zero_kernel<<<(nw + 255) / 256, 256, 0, stream>>>((unsigned int*)hbf0, nw);
  }
  // 1) convert x -> bf16; transpose weights -> [N][K] bf16
  {
    const int nx = 64 * 1024 * 256;
    f32_to_bf16_kernel<<<nx / 256, 256, 0, stream>>>(x, xb, nx);
    wtrans_kernel<<<(256 * 1536 + 255) / 256, 256, 0, stream>>>(W_i0, WiT0, 256, 1536);
    wtrans_kernel<<<(512 * 1536 + 255) / 256, 256, 0, stream>>>(W_h0, WhT0, 512, 1536);
    wtrans_kernel<<<(512 * 1536 + 255) / 256, 256, 0, stream>>>(W_i1, WiT1, 512, 1536);
    wtrans_kernel<<<(512 * 1536 + 255) / 256, 256, 0, stream>>>(W_h1, WhT1, 512, 1536);
    wtrans_kernel<<<(512 * 512 + 255) / 256, 256, 0, stream>>>(W_out, WoT, 512, 512);
  }
  // 2) layer 0: xg = x @ W_i0 + b_i0 ; scan (writes y0 bf16 + c0)
  gemm_xg_kernel<<<dim3(512, 12), 256, 0, stream>>>(xb, WiT0, b_i0, xg, 256);
  gru_scan_kernel<<<32, 128, SCAN_LDS, stream>>>(
      xg, WhT0, b_hn0, hbf0, yb, out + 64 * 512, nullptr, ctr0, 1024);
  // 3) layer 1: xg = y0 @ W_i1 + b_i1 ; scan (writes c1 + tanh(c1) bf16)
  gemm_xg_kernel<<<dim3(512, 12), 256, 0, stream>>>(yb, WiT1, b_i1, xg, 512);
  gru_scan_kernel<<<32, 128, SCAN_LDS, stream>>>(
      xg, WhT1, b_hn1, hbf1, nullptr, out + 2 * 64 * 512, c1t, ctr1, 1024);
  // 4) dense head
  dense_out_kernel<<<32, 128, 0, stream>>>(c1t, WoT, b_out, out);
}
